// BioDetailedSTDPSNN_64123861729691
// MI455X (gfx1250) — compile-verified, matrix operand order checked by reading, not run on hardware
//
#include <hip/hip_runtime.h>

// ---------------- problem constants (mirror reference) ----------------
constexpr int T    = 512;
constexpr int B    = 256;
constexpr int DIN  = 512;
constexpr int DH   = 512;
constexpr int DOUT = 128;

constexpr float BETA    = 0.9f;
constexpr float STDP_LR = 0.05f;
constexpr float ADAPT   = 0.02f;
constexpr float TARGET  = 0.1f;

constexpr int G   = 16;         // batch groups (independent recurrences)
constexpr int BT  = 16;         // batch rows per group (one WMMA M tile)
constexpr int NT2 = DH / 16;    // 32  N-tiles for s@W_post
constexpr int KC2 = DH / 32;    // 16  K-chunks of 32 (bf16 WMMA K)
constexpr int NTH = DOUT / 16;  // 8   N-tiles for s@W_head

// ---------------- ws layout (bytes) ----------------
constexpr size_t OFF_PRE    = 0;           // f32 [T*B, DH]
constexpr size_t OFF_RAWLOG = 268435456;   // f32 [T*B, DOUT]
constexpr size_t OFF_WPK2   = 335544320;   // bf16 packed W_post frags
constexpr size_t OFF_WPKH   = 336068608;   // bf16 packed W_head frags
constexpr size_t OFF_PSTDP  = 336199680;   // f32 [T*G]
constexpr size_t OFF_PS     = 336232448;   // f32 [T*G]
constexpr size_t OFF_PCTH   = 336265216;   // f32 [T*G]
constexpr size_t OFF_SSIG   = 336297984;   // f32 [T]

typedef __attribute__((ext_vector_type(2)))  float        v2f;
typedef __attribute__((ext_vector_type(8)))  float        v8f;
typedef __attribute__((ext_vector_type(16))) __bf16       v16bf;
typedef __attribute__((ext_vector_type(8)))  unsigned int v8u;
typedef __attribute__((ext_vector_type(4)))  unsigned int v4u;
typedef __attribute__((ext_vector_type(4)))  int          v4i;
typedef __attribute__((ext_vector_type(8)))  int          v8i;

// Fragment index map for 16-bit 16x16x32 WMMA operands (ISA 7.12.2):
//   value v (0..15), lane l (0..31)
//   K(v,l) = (v>=8?16:0) + (l>=16?8:0) + ((v>>1)&3)*2 + (v&1)
//   A: M = l&15 ; B: N = l&15
__device__ __forceinline__ int frag_k(int v, int l) {
  return ((v >> 3) << 4) + ((l >> 4) << 3) + (((v >> 1) & 3) << 1) + (v & 1);
}

// -------------------------------------------------------------------
// Tensor Data Mover: 2D tile load Global->LDS (D# per ISA ch.8).
// data_size=4B; optional LDS row padding via D#.pad_* so padded LDS
// strides (33 / 260 dwords) come straight out of the DMA engine.
// lds_byte_off is relative to the dynamic-LDS base (kernels using this
// declare no static __shared__, so dynamic LDS starts at offset 0).
// -------------------------------------------------------------------
__device__ __forceinline__ void tdm_load_2d(unsigned lds_byte_off,
                                            const void* gptr,
                                            unsigned tile_d0, unsigned tile_d1,
                                            unsigned long long stride0_elems,
                                            unsigned pad_interval_code,
                                            unsigned pad_amount_code) {
  unsigned long long ga = (unsigned long long)gptr;
  v4u g0;
  g0[0] = 1u;                                  // count=1, user descriptor
  g0[1] = lds_byte_off;                        // lds_addr
  g0[2] = (unsigned)ga;                        // global_addr[31:0]
  g0[3] = ((unsigned)(ga >> 32) & 0x01FFFFFFu) // global_addr[56:32]
          | (2u << 30);                        // type=2 ("image")
  // group1: mask=0, data_size=2(4B), pad_enable, pad codes
  unsigned d0 = (2u << 16) | (1u << 20) | (pad_interval_code << 22) |
                (pad_amount_code << 25);
  unsigned td0 = tile_d0, td1 = tile_d1;       // tensor dims == tile dims
  unsigned d1 = (td0 & 0xFFFFu) << 16;                       // bits[79:48] lo
  unsigned d2 = (td0 >> 16) | ((td1 & 0xFFFFu) << 16);       // dim0 hi, dim1 lo
  unsigned d3 = (td1 >> 16) | ((tile_d0 & 0xFFFFu) << 16);   // dim1 hi, tile0
  unsigned d4 = (tile_d1 & 0xFFFFu);                         // tile1, tile2=0
  unsigned d5 = (unsigned)(stride0_elems & 0xFFFFFFFFull);   // dim0_stride lo
  unsigned d6 = (unsigned)((stride0_elems >> 32) & 0xFFFFull);
  unsigned d7 = 0u;                                          // dim1_stride=0
  v8i g1;
  g1[0] = (int)d0; g1[1] = (int)d1; g1[2] = (int)d2; g1[3] = (int)d3;
  g1[4] = (int)d4; g1[5] = (int)d5; g1[6] = (int)d6; g1[7] = (int)d7;
  v4i z4; z4[0] = 0; z4[1] = 0; z4[2] = 0; z4[3] = 0;
#if defined(__clang_major__) && (__clang_major__ >= 23)
  v8i z8; z8[0]=0; z8[1]=0; z8[2]=0; z8[3]=0; z8[4]=0; z8[5]=0; z8[6]=0; z8[7]=0;
  __builtin_amdgcn_tensor_load_to_lds(g0, g1, z4, z4, z8, 0);
#else
  __builtin_amdgcn_tensor_load_to_lds(g0, g1, z4, z4, 0);
#endif
}

// -------------------------------------------------------------------
// Phase 0: pack W_post / W_head (fp32) into bf16 WMMA-B fragment order.
// -------------------------------------------------------------------
__global__ void snn_pack_weights(const float* __restrict__ Wp,
                                 const float* __restrict__ Wh,
                                 unsigned short* __restrict__ WPK2u,
                                 unsigned short* __restrict__ WPKHu) {
  int idx = blockIdx.x * blockDim.x + threadIdx.x;
  const int total2 = KC2 * NT2 * 512;   // 262144
  const int totalH = KC2 * NTH * 512;   // 65536
  __bf16* WPK2 = (__bf16*)WPK2u;
  __bf16* WPKH = (__bf16*)WPKHu;
  if (idx < total2) {
    int v = idx & 15, l = (idx >> 4) & 31, f = idx >> 9;
    int nt = f % NT2, kc = f / NT2;
    int K = kc * 32 + frag_k(v, l);
    int N = nt * 16 + (l & 15);
    WPK2[idx] = (__bf16)Wp[(size_t)K * DH + N];
  } else if (idx < total2 + totalH) {
    int j = idx - total2;
    int v = j & 15, l = (j >> 4) & 31, f = j >> 9;
    int nt = f % NTH, kc = f / NTH;
    int K = kc * 32 + frag_k(v, l);
    int N = nt * 16 + (l & 15);
    WPKH[j] = (__bf16)Wh[(size_t)K * DOUT + N];
  }
}

// -------------------------------------------------------------------
// Phase 1: PRE = x_seq @ W_in + b_in  (fp32 WMMA 16x16x4).
// 512 thr (16 waves), tile M=64 x N=256, K-slab=32.
// TDM double-buffering: each wave DMAs its slice of the next slab
// (A: 4 rows x 32, B: 2 rows x 256) while computing the current one;
// sync = s_wait_tensorcnt 0 + workgroup barrier.
// -------------------------------------------------------------------
__global__ void __launch_bounds__(512, 1)
snn_pre_gemm(const float* __restrict__ x, const float* __restrict__ Win,
             const float* __restrict__ bin, float* __restrict__ PRE) {
  extern __shared__ float sm[];
  constexpr int ASTR = 33, BSTR = 260;
  constexpr int ABUF = 64 * ASTR;           // 2112 floats
  constexpr int BBUF = 32 * BSTR;           // 8320 floats
  constexpr int BUF  = ABUF + BBUF;         // per parity buffer

  int tid = threadIdx.x, lane = tid & 31, wid = tid >> 5;
  int mb = blockIdx.x >> 1, nb = blockIdx.x & 1;
  int row0 = mb * 64, n0 = nb * 256;
  int lh = lane >> 4, ln = lane & 15;
  int mt  = wid >> 2;        // 0..3
  int nt0 = (wid & 3) * 4;   // 0,4,8,12

  // wave-sliced TDM issue for slab at k-offset ks into parity buffer p
  auto issue_slab = [&](int p, int ks) {
    // A: x rows [row0+4w, +4), cols [ks, +32); LDS rows stride 33 (pad 1/32DW)
    tdm_load_2d((unsigned)((p * BUF + (4 * wid) * ASTR) * 4),
                x + (size_t)(row0 + 4 * wid) * DIN + ks, 32u, 4u,
                (unsigned long long)DIN, 4u /*32 DW*/, 0u /*+1 DW*/);
    // B: W_in rows [ks+2w, +2), cols [n0, +256); stride 260 (pad 4/256DW)
    tdm_load_2d((unsigned)((p * BUF + ABUF + (2 * wid) * BSTR) * 4),
                Win + (size_t)(ks + 2 * wid) * DH + n0, 256u, 2u,
                (unsigned long long)DH, 7u /*256 DW*/, 3u /*+4 DW*/);
  };

  v8f acc[4];
#pragma unroll
  for (int q = 0; q < 4; ++q)
#pragma unroll
    for (int v = 0; v < 8; ++v) acc[q][v] = 0.f;

  issue_slab(0, 0);
  __builtin_amdgcn_s_wait_tensorcnt(0);
  __syncthreads();

  for (int s = 0; s < DIN / 32; ++s) {
    int p = s & 1;
    if (s + 1 < DIN / 32) issue_slab(p ^ 1, (s + 1) * 32);

    const float* As = sm + p * BUF;
    const float* Bs = sm + p * BUF + ABUF;
#pragma unroll
    for (int kk = 0; kk < 32; kk += 4) {
      int ka = kk + 2 * lh;
      v2f a;
      a[0] = As[(mt * 16 + ln) * ASTR + ka];
      a[1] = As[(mt * 16 + ln) * ASTR + ka + 1];
#pragma unroll
      for (int q = 0; q < 4; ++q) {
        int nn = (nt0 + q) * 16 + ln;
        v2f b;
        b[0] = Bs[ka * BSTR + nn];
        b[1] = Bs[(ka + 1) * BSTR + nn];
        acc[q] = __builtin_amdgcn_wmma_f32_16x16x4_f32(
            false, a, false, b, (short)0, acc[q], false, false);
      }
    }
    __builtin_amdgcn_s_wait_tensorcnt(0);  // next slab landed
    __syncthreads();                       // everyone done reading slab s
  }

#pragma unroll
  for (int q = 0; q < 4; ++q) {
    int nn = n0 + (nt0 + q) * 16 + ln;
    float bv = bin[nn];
#pragma unroll
    for (int v = 0; v < 8; ++v) {
      int m = mt * 16 + v + 8 * lh;
      PRE[(size_t)(row0 + m) * DH + nn] = acc[q][v] + bv;
    }
  }
}

// -------------------------------------------------------------------
// Phase 2: sequential recurrence, 16 independent workgroups.
// bf16 WMMA for s_t@W_post / s_t@W_head (s_t exact in bf16).
// -------------------------------------------------------------------
__device__ __forceinline__ v16bf load_afrag(const unsigned short* SBFu, int ln,
                                            int lh, int kc) {
  // 8 dword reads (each = one aligned bf16 K-pair), bit-cast to v16bf
  v8u w;
  int kb = kc * 32 + (lh ? 8 : 0);
#pragma unroll
  for (int gg = 0; gg < 8; ++gg) {
    int K = kb + ((gg & 4) ? 16 : 0) + (gg & 3) * 2;
    w[gg] = *(const unsigned int*)(SBFu + ln * DH + K);
  }
  return __builtin_bit_cast(v16bf, w);
}

__global__ void __launch_bounds__(512, 1)
snn_recurrence(const float* __restrict__ PRE,
               const unsigned short* __restrict__ WPK2u,
               const float* __restrict__ bpost,
               const unsigned short* __restrict__ WPKHu,
               const float* __restrict__ bhead,
               float* __restrict__ RAWLOG, float* __restrict__ PSTDP,
               float* __restrict__ PS, float* __restrict__ PCTH) {
  extern __shared__ char smem[];
  float* V    = (float*)smem;             // [16][DH]
  float* CPR  = V + 16 * DH;              // [16][DH]
  float* CPO  = CPR + 16 * DH;            // [16][DH]
  float* HAC  = CPO + 16 * DH;            // [16][256] head partial frags
  float* ROWS = HAC + 16 * 256;           // [16] row spike sums
  float* ROWC = ROWS + 16;                // [16] row c_th
  float* WST  = ROWC + 16;                // [16] wave stdp partials
  float* CTH  = WST + 16;                 // [16]
  unsigned short* SBFu = (unsigned short*)(CTH + 16);  // [16][DH] bf16 spikes
  __bf16* SBF = (__bf16*)SBFu;

  const __bf16* WPK2 = (const __bf16*)WPK2u;
  const __bf16* WPKH = (const __bf16*)WPKHu;

  int tid = threadIdx.x, lane = tid & 31, wid = tid >> 5;
  int lh = lane >> 4, ln = lane & 15;
  int g = blockIdx.x, b0 = g * BT;

  for (int i = tid; i < 16 * DH; i += 512) { V[i] = 0.f; CPR[i] = 0.f; CPO[i] = 0.f; }
  if (tid < 16) CTH[tid] = 0.f;
  __syncthreads();

  for (int t = 0; t < T; ++t) {
    // ---- stage 1: LIF elementwise; wave w owns batch row m=w ----
    int m = wid;
    const float* prerow = PRE + ((size_t)t * B + b0 + m) * DH;
    {
      int tn = (t + 1 < T) ? (t + 1) : t;
      __builtin_prefetch(PRE + ((size_t)tn * B + b0 + m) * DH + lane * 16, 0, 0);
    }
    float cth_old = CTH[m];
    float ssum = 0.f;
#pragma unroll
    for (int i = 0; i < DH / 32; ++i) {
      int h = lane + 32 * i;
      float pre = prerow[h];
      float v = BETA * V[m * DH + h] + (pre - cth_old);
      float s = (v - 1.0f) > 0.f ? 1.f : 0.f;
      v -= s;  // soft reset (THRESHOLD = 1)
      V[m * DH + h] = v;
      CPR[m * DH + h] = 0.95f * CPR[m * DH + h] + 0.05f * pre;
      SBF[m * DH + h] = (__bf16)s;
      ssum += s;
    }
#pragma unroll
    for (int off = 16; off > 0; off >>= 1) ssum += __shfl_down(ssum, off, 32);
    if (lane == 0) {
      float cth_new = 0.95f * cth_old + ADAPT * (ssum * (1.f / DH) - TARGET);
      CTH[m] = cth_new;
      ROWS[m] = ssum;
      ROWC[m] = cth_new;
    }
    __syncthreads();

    // ---- stage 2: bf16 WMMA GEMMs (uniform trips, EXEC stays full) ----
    int nt0 = wid * 2, nt1 = wid * 2 + 1;
    v8f acc0, acc1, acch;
#pragma unroll
    for (int v = 0; v < 8; ++v) { acc0[v] = 0.f; acc1[v] = 0.f; acch[v] = 0.f; }

    for (int kc = 0; kc < KC2; ++kc) {
      v16bf a = load_afrag(SBFu, ln, lh, kc);
      const v16bf* bp0 =
          (const v16bf*)(WPK2 + ((size_t)(kc * NT2 + nt0) * 32 + lane) * 16);
      const v16bf* bp1 =
          (const v16bf*)(WPK2 + ((size_t)(kc * NT2 + nt1) * 32 + lane) * 16);
      acc0 = __builtin_amdgcn_wmma_f32_16x16x32_bf16(false, a, false, *bp0,
                                                     (short)0, acc0, false, false);
      acc1 = __builtin_amdgcn_wmma_f32_16x16x32_bf16(false, a, false, *bp1,
                                                     (short)0, acc1, false, false);
    }
    int hnt = wid & 7, hks = (wid >> 3) * 8;
    for (int kc = hks; kc < hks + 8; ++kc) {
      v16bf a = load_afrag(SBFu, ln, lh, kc);
      const v16bf* bh =
          (const v16bf*)(WPKH + ((size_t)(kc * NTH + hnt) * 32 + lane) * 16);
      acch = __builtin_amdgcn_wmma_f32_16x16x32_bf16(false, a, false, *bh,
                                                     (short)0, acch, false, false);
    }

    // c_post EMA + stdp partial (fragment layout: M=v+8*lh, N=l&15)
    float stdp = 0.f;
    {
      int n0g = nt0 * 16 + ln, n1g = nt1 * 16 + ln;
      float bp0v = bpost[n0g], bp1v = bpost[n1g];
#pragma unroll
      for (int v = 0; v < 8; ++v) {
        int mm = v + 8 * lh;
        float c0 = 0.95f * CPO[mm * DH + n0g] + 0.05f * (acc0[v] + bp0v);
        CPO[mm * DH + n0g] = c0;
        stdp += CPR[mm * DH + n0g] * c0;
        float c1 = 0.95f * CPO[mm * DH + n1g] + 0.05f * (acc1[v] + bp1v);
        CPO[mm * DH + n1g] = c1;
        stdp += CPR[mm * DH + n1g] * c1;
      }
    }
#pragma unroll
    for (int off = 16; off > 0; off >>= 1) stdp += __shfl_down(stdp, off, 32);
    if (lane == 0) WST[wid] = stdp;
#pragma unroll
    for (int v = 0; v < 8; ++v) HAC[wid * 256 + v * 32 + lane] = acch[v];
    __syncthreads();

    // ---- stage 3: partials + head combine + raw logits ----
    if (tid == 0) {
      float a = 0.f, bs = 0.f, c = 0.f;
      for (int i = 0; i < 16; ++i) { a += WST[i]; bs += ROWS[i]; c += ROWC[i]; }
      PSTDP[t * G + g] = a;
      PS[t * G + g] = bs;
      PCTH[t * G + g] = c;
    }
    for (int i = tid; i < 16 * DOUT; i += 512) {
      int mm = i >> 7, o = i & 127;
      int ntile = o >> 4, N = o & 15;
      int l = (mm >> 3) * 16 + N, v = mm & 7;
      float val = HAC[ntile * 256 + v * 32 + l] +
                  HAC[(ntile + 8) * 256 + v * 32 + l] + bhead[o];
      RAWLOG[((size_t)t * B + b0 + mm) * DOUT + o] = val;
    }
    __syncthreads();
  }
}

// -------------------------------------------------------------------
// Phase 3a: stdp_signal[t] + the three scalar outputs.
// -------------------------------------------------------------------
__global__ void snn_reduce(const float* __restrict__ PSTDP,
                           const float* __restrict__ PS,
                           const float* __restrict__ PCTH,
                           float* __restrict__ SSIG,
                           float* __restrict__ outTail) {
  __shared__ float s1[T], s2[T], s3[T];
  int t = threadIdx.x;
  float a = 0.f, b = 0.f, c = 0.f;
  for (int gg = 0; gg < G; ++gg) {
    a += PSTDP[t * G + gg];
    b += PS[t * G + gg];
    c += PCTH[t * G + gg];
  }
  float sig = a * (1.f / ((float)B * DH));
  SSIG[t] = sig;
  s1[t] = sig; s2[t] = b; s3[t] = c;
  __syncthreads();
  if (t == 0) {
    float x1 = 0.f, x2 = 0.f, x3 = 0.f;
    for (int i = 0; i < T; ++i) { x1 += s1[i]; x2 += s2[i]; x3 += s3[i]; }
    outTail[0] = x2 * (1.f / ((float)T * B * DH));  // mean spike rate
    outTail[1] = x1 * (1.f / (float)T);             // mean stdp signal
    outTail[2] = x3 * (1.f / ((float)T * B));       // mean threshold
  }
}

// -------------------------------------------------------------------
// Phase 3b: logits_seq = raw + STDP_LR*sig[t]; out = mean over T.
// -------------------------------------------------------------------
__global__ void snn_finalize(const float* __restrict__ RAWLOG,
                             const float* __restrict__ SSIG,
                             float* __restrict__ OUT,
                             float* __restrict__ LOGITS) {
  __shared__ float sig[T];
  for (int i = threadIdx.x; i < T; i += blockDim.x) sig[i] = SSIG[i];
  __syncthreads();
  int idx = blockIdx.x * blockDim.x + threadIdx.x;  // 0..B*DOUT-1
  float acc = 0.f;
  for (int t = 0; t < T; ++t) {
    float v = RAWLOG[(size_t)t * B * DOUT + idx] + STDP_LR * sig[t];
    LOGITS[(size_t)t * B * DOUT + idx] = v;
    acc += v;
  }
  OUT[idx] = acc * (1.f / (float)T);
}

// -------------------------------------------------------------------
extern "C" void kernel_launch(void* const* d_in, const int* in_sizes, int n_in,
                              void* d_out, int out_size, void* d_ws,
                              size_t ws_size, hipStream_t stream) {
  (void)in_sizes; (void)n_in; (void)out_size; (void)ws_size;
  const float* x     = (const float*)d_in[0];
  const float* Win   = (const float*)d_in[1];
  const float* bin   = (const float*)d_in[2];
  const float* Wpost = (const float*)d_in[3];
  const float* bpost = (const float*)d_in[4];
  const float* Whead = (const float*)d_in[5];
  const float* bhead = (const float*)d_in[6];

  char* ws = (char*)d_ws;
  float*          PRE    = (float*)(ws + OFF_PRE);
  float*          RAWLOG = (float*)(ws + OFF_RAWLOG);
  unsigned short* WPK2   = (unsigned short*)(ws + OFF_WPK2);
  unsigned short* WPKH   = (unsigned short*)(ws + OFF_WPKH);
  float*          PSTDP  = (float*)(ws + OFF_PSTDP);
  float*          PS     = (float*)(ws + OFF_PS);
  float*          PCTH   = (float*)(ws + OFF_PCTH);
  float*          SSIG   = (float*)(ws + OFF_SSIG);

  float* OUT     = (float*)d_out;                  // [B,DOUT]
  float* LOGITS  = OUT + B * DOUT;                 // [T,B,DOUT]
  float* OUTTAIL = LOGITS + (size_t)T * B * DOUT;  // sr, stdp, th

  // Phase 0: weight packing
  snn_pack_weights<<<1280, 256, 0, stream>>>(Wpost, Whead, WPK2, WPKH);

  // Phase 1: pre-activation GEMM (TDM double-buffered slabs)
  size_t smem1 = (size_t)2 * (64 * 33 + 32 * 260) * sizeof(float);  // 83,456 B
  snn_pre_gemm<<<(T * B / 64) * 2, 512, smem1, stream>>>(x, Win, bin, PRE);

  // Phase 2: 16 independent recurrences
  size_t smem2 = (size_t)(3 * 16 * DH + 16 * 256 + 4 * 16) * sizeof(float) +
                 (size_t)16 * DH * sizeof(unsigned short);
  snn_recurrence<<<G, 512, smem2, stream>>>(PRE, WPK2, bpost, WPKH, bhead,
                                            RAWLOG, PSTDP, PS, PCTH);

  // Phase 3: global stdp signal + scalars, then logits/out
  snn_reduce<<<1, T, 0, stream>>>(PSTDP, PS, PCTH, SSIG, OUTTAIL);
  snn_finalize<<<(B * DOUT) / 256, 256, 0, stream>>>(RAWLOG, SSIG, OUT, LOGITS);
}
